// BG_LSTM_18726057410809
// MI455X (gfx1250) — matrix operand order repaired and checked by reading, not compiled, not run on hardware
//
#include <hip/hip_runtime.h>

#define BATCH    512
#define SEQT     512
#define HID      256
#define G4       1024
#define MTILE    16
#define LDH      264          // padded bf16 row stride for h in LDS (528B, 16B-aligned rows)
#define LDW      136          // padded bf16 row stride for LDS weight half (272B: 16B-aligned, 4-bank row skew)
#define NTHREADS 512          // 16 wave32 waves
#define KRES     128          // K range covered by register-resident weights

typedef __attribute__((ext_vector_type(16))) __bf16 v16bf;
typedef __attribute__((ext_vector_type(8)))  float  v8f;

union BFrag {
    uint4 u[2];
    v16bf v;
};

__device__ __forceinline__ unsigned short f32_to_bf16_bits(float f) {
    union { float f; unsigned int u; } in;
    in.f = f;
    unsigned int u = in.u;
    u += 0x7FFFu + ((u >> 16) & 1u);   // round-to-nearest-even
    return (unsigned short)(u >> 16);
}

__device__ __forceinline__ float fast_sigmoid(float z) {
    return 1.0f / (1.0f + __expf(-z));
}
__device__ __forceinline__ float fast_tanh(float z) {
    return 2.0f / (1.0f + __expf(-2.0f * z)) - 1.0f;
}

// Prep: W_hh fp32 [4H][H] row-major -> bf16 bits, same layout, into workspace.
__global__ void whh_to_bf16_kernel(const float* __restrict__ src,
                                   unsigned short* __restrict__ dst, int n) {
    int i = blockIdx.x * blockDim.x + threadIdx.x;
    if (i < n) dst[i] = f32_to_bf16_bits(src[i]);
}

__global__ __launch_bounds__(NTHREADS) void lstm_wmma_kernel(
    const float* __restrict__ x,            // [B][T]
    const float* __restrict__ W_ih,         // [4H]
    const unsigned short* __restrict__ Wb,  // bf16 bits of W_hh [4H][H]
    const float* __restrict__ b_ih,         // [4H]
    const float* __restrict__ b_hh,         // [4H]
    const float* __restrict__ W_fc,         // [H]
    const float* __restrict__ b_fc,         // [1]
    float* __restrict__ out)                // [B]
{
    // K = 128..255 half of W_hh, resident in LDS for the whole kernel (272 KB)
    __shared__ unsigned short wS[G4 * LDW];
    __shared__ unsigned short hS[MTILE * LDH];  // h (bf16) [16][LDH]
    __shared__ float xS[MTILE];
    __shared__ float outAcc[MTILE];

    const int tid  = threadIdx.x;
    const int wave = tid >> 5;
    const int lane = tid & 31;
    const int hsel = lane >> 4;            // half-wave select
    const int l16  = lane & 15;
    const int b0   = blockIdx.x * MTILE;   // batch-tile base row
    const int jcol = (wave << 4) + l16;    // hidden column this lane owns in C tiles

    // init h = 0, output accumulator = 0
    for (int i = tid; i < MTILE * LDH; i += NTHREADS) hS[i] = 0;
    if (tid < MTILE) outAcc[tid] = 0.0f;

    // Cooperative preload of the K=128..255 weight half into LDS (uint4 chunks).
    for (int i = tid; i < G4 * 16; i += NTHREADS) {
        int r = i >> 4;       // weight row
        int c = i & 15;       // 8-element chunk within the 128-element half-row
        *reinterpret_cast<uint4*>(wS + r * LDW + c * 8) =
            *reinterpret_cast<const uint4*>(Wb + r * HID + KRES + c * 8);
    }

    // Per-lane gate coefficients (constant over time): x-weight and combined bias
    float wI[4], bb[4];
#pragma unroll
    for (int g = 0; g < 4; ++g) {
        int j = g * HID + jcol;
        wI[g] = W_ih[j];
        bb[g] = b_ih[j] + b_hh[j];
    }
    const float wfc = W_fc[jcol];

    // B-fragment layout (32x16 bf16): lanes 0-15 -> col N=l16, K=kk..kk+15 contiguous;
    //                                 lanes 16-31 -> K=kk+16..kk+31.
    // Register-resident weight fragments for K = 0..127  (4 x 4 x 8 = 128 VGPRs)
    BFrag bres[4][4];
#pragma unroll
    for (int g = 0; g < 4; ++g) {
        const unsigned short* wrow = Wb + (size_t)(g * HID + jcol) * HID + hsel * 16;
#pragma unroll
        for (int k4 = 0; k4 < 4; ++k4) {
            const uint4* bp = reinterpret_cast<const uint4*>(wrow + k4 * 32);
            bres[g][k4].u[0] = bp[0];
            bres[g][k4].u[1] = bp[1];
        }
    }

    // LDS base pointers for the K=128..255 weight half
    const unsigned short* wlds[4];
#pragma unroll
    for (int g = 0; g < 4; ++g)
        wlds[g] = wS + (g * HID + jcol) * LDW + hsel * 16;

    // Cell state lives in registers: element r -> (row r + 8*hsel, col jcol)
    float cst[8], hlast[8];
#pragma unroll
    for (int r = 0; r < 8; ++r) { cst[r] = 0.0f; hlast[r] = 0.0f; }

    // A layout (16x32 bf16): lanes 0-15 -> row M=l16, K = kk+0..15;
    //                        lanes 16-31 -> K = kk+16..31.
    const unsigned short* aBase = hS + l16 * LDH + hsel * 16;

    __syncthreads();   // wS preload + hS init complete

    for (int t = 0; t < SEQT; ++t) {
        if (tid < MTILE) xS[tid] = x[(size_t)(b0 + tid) * SEQT + t];
        __syncthreads();   // xS ready; previous step's hS writes visible

        // Launder a zero offset so the LDS weight loads below cannot be hoisted
        // out of the time loop by LICM (which would force spills).
        int off = 0;
        asm volatile("" : "+v"(off));

        v8f acc0 = {}, acc1 = {}, acc2 = {}, acc3 = {};

        // ---- register-resident half: K = 0..127 ----
#pragma unroll
        for (int k4 = 0; k4 < 4; ++k4) {
            BFrag a;
            const uint4* ap = reinterpret_cast<const uint4*>(aBase + k4 * 32);
            a.u[0] = ap[0];   // K = kk + hsel*16 + 0..7
            a.u[1] = ap[1];   // K = kk + hsel*16 + 8..15
            acc0 = __builtin_amdgcn_wmma_f32_16x16x32_bf16(false, a.v, false, bres[0][k4].v,
                                                           (short)0, acc0, false, false);
            acc1 = __builtin_amdgcn_wmma_f32_16x16x32_bf16(false, a.v, false, bres[1][k4].v,
                                                           (short)0, acc1, false, false);
            acc2 = __builtin_amdgcn_wmma_f32_16x16x32_bf16(false, a.v, false, bres[2][k4].v,
                                                           (short)0, acc2, false, false);
            acc3 = __builtin_amdgcn_wmma_f32_16x16x32_bf16(false, a.v, false, bres[3][k4].v,
                                                           (short)0, acc3, false, false);
        }

        // ---- LDS-resident half: K = 128..255 ----
#pragma unroll
        for (int k4 = 0; k4 < 4; ++k4) {
            BFrag a;
            const uint4* ap = reinterpret_cast<const uint4*>(aBase + KRES + k4 * 32);
            a.u[0] = ap[0];
            a.u[1] = ap[1];

            BFrag s0, s1, s2, s3;
            const uint4* q0 = reinterpret_cast<const uint4*>(wlds[0] + k4 * 32 + off);
            const uint4* q1 = reinterpret_cast<const uint4*>(wlds[1] + k4 * 32 + off);
            const uint4* q2 = reinterpret_cast<const uint4*>(wlds[2] + k4 * 32 + off);
            const uint4* q3 = reinterpret_cast<const uint4*>(wlds[3] + k4 * 32 + off);
            s0.u[0] = q0[0]; s0.u[1] = q0[1];
            s1.u[0] = q1[0]; s1.u[1] = q1[1];
            s2.u[0] = q2[0]; s2.u[1] = q2[1];
            s3.u[0] = q3[0]; s3.u[1] = q3[1];

            acc0 = __builtin_amdgcn_wmma_f32_16x16x32_bf16(false, a.v, false, s0.v,
                                                           (short)0, acc0, false, false);
            acc1 = __builtin_amdgcn_wmma_f32_16x16x32_bf16(false, a.v, false, s1.v,
                                                           (short)0, acc1, false, false);
            acc2 = __builtin_amdgcn_wmma_f32_16x16x32_bf16(false, a.v, false, s2.v,
                                                           (short)0, acc2, false, false);
            acc3 = __builtin_amdgcn_wmma_f32_16x16x32_bf16(false, a.v, false, s3.v,
                                                           (short)0, acc3, false, false);
        }

        __syncthreads();   // all waves finished reading hS for this step

        // Pointwise LSTM update fully in registers; write new h (bf16) to LDS.
#pragma unroll
        for (int r = 0; r < 8; ++r) {
            float xv = xS[r + 8 * hsel];
            float zi = acc0[r] + xv * wI[0] + bb[0];
            float zf = acc1[r] + xv * wI[1] + bb[1];
            float zg = acc2[r] + xv * wI[2] + bb[2];
            float zo = acc3[r] + xv * wI[3] + bb[3];
            float ig = fast_sigmoid(zi);
            float fg = fast_sigmoid(zf);
            float gg = fast_tanh(zg);
            float og = fast_sigmoid(zo);
            float c  = fg * cst[r] + ig * gg;
            cst[r]   = c;
            float h  = og * fast_tanh(c);
            hlast[r] = h;
            hS[(r + 8 * hsel) * LDH + jcol] = f32_to_bf16_bits(h);
        }
    }

    // FC head: out[b] = sum_j relu(h[b][j]) * W_fc[j] + b_fc
    __syncthreads();
#pragma unroll
    for (int r = 0; r < 8; ++r) {
        float h = hlast[r];
        float rl = h > 0.0f ? h : 0.0f;
        atomicAdd(&outAcc[r + 8 * hsel], rl * wfc);   // ds_add_f32
    }
    __syncthreads();
    if (tid < MTILE) out[b0 + tid] = outAcc[tid] + b_fc[0];
}

extern "C" void kernel_launch(void* const* d_in, const int* in_sizes, int n_in,
                              void* d_out, int out_size, void* d_ws, size_t ws_size,
                              hipStream_t stream) {
    const float* x    = (const float*)d_in[0];
    const float* W_ih = (const float*)d_in[1];
    const float* W_hh = (const float*)d_in[2];
    const float* b_ih = (const float*)d_in[3];
    const float* b_hh = (const float*)d_in[4];
    const float* W_fc = (const float*)d_in[5];
    const float* b_fc = (const float*)d_in[6];
    float* out        = (float*)d_out;

    unsigned short* Wb = (unsigned short*)d_ws;   // bf16 copy of W_hh (512 KB)

    const int nW = G4 * HID;
    whh_to_bf16_kernel<<<nW / 256, 256, 0, stream>>>(W_hh, Wb, nW);
    lstm_wmma_kernel<<<BATCH / MTILE, NTHREADS, 0, stream>>>(
        x, W_ih, Wb, b_ih, b_hh, W_fc, b_fc, out);
}
